// GroupedQueryAttentionWithRoPE_45002667327640
// MI455X (gfx1250) — matrix operand order repaired and checked by reading, not compiled, not run on hardware
//
#include <hip/hip_runtime.h>
#include <stdint.h>
#include <stddef.h>

// ---------------------------------------------------------------------------
// CDNA5 (gfx1250) WMMA types
// ---------------------------------------------------------------------------
typedef __attribute__((ext_vector_type(16))) __bf16 v16bf;
typedef __attribute__((ext_vector_type(8)))  float  v8f;

union Frag16 { v16bf v; uint32_t u[8]; };

// ---------------------------------------------------------------------------
// Async global->LDS (CDNA5 GLOBAL_LOAD_ASYNC_TO_LDS_B128, ASYNCcnt-tracked).
// Probe result (round 2 diagnostic): param 0 is GCC-vector `int v4` pointer in
// the __device__ (global) address space; 4 args total. Clang implicitly
// addrspace-casts generic pointers for builtin args in HIP mode.
// ---------------------------------------------------------------------------
#if defined(__has_builtin)
#if __has_builtin(__builtin_amdgcn_global_load_async_to_lds_b128)
#define HAVE_ASYNC_LDS 1
#endif
#if __has_builtin(__builtin_amdgcn_s_wait_asynccnt)
#define HAVE_WAIT_ASYNC 1
#endif
#endif

typedef int v4i_gcc __attribute__((vector_size(16)));

__device__ __forceinline__ void async_copy16(const uint16_t* g, uint16_t* l) {
#if defined(HAVE_ASYNC_LDS)
  __builtin_amdgcn_global_load_async_to_lds_b128(
      (v4i_gcc*)g, (v4i_gcc*)l, /*imm offset*/ 0, /*cpol*/ 0);
#else
  *(uint4*)l = *(const uint4*)g;
#endif
}

__device__ __forceinline__ void async_wait_all() {
#if defined(HAVE_ASYNC_LDS)
#if defined(HAVE_WAIT_ASYNC)
  __builtin_amdgcn_s_wait_asynccnt(0);
#else
  asm volatile("s_wait_asynccnt 0x0" ::: "memory");
#endif
#endif
}

__device__ __forceinline__ uint16_t f2bf(float f) {
  uint32_t u = __float_as_uint(f);
  uint32_t r = (u + 0x7FFFu + ((u >> 16) & 1u)) >> 16;  // round-nearest-even
  return (uint16_t)r;
}

// ---------------------------------------------------------------------------
// Elementwise prep kernels
// ---------------------------------------------------------------------------
__global__ void cvt_bf16_kernel(const float* __restrict__ in,
                                uint16_t* __restrict__ out, int n) {
  int i = blockIdx.x * blockDim.x + threadIdx.x;
  if (i < n) out[i] = f2bf(in[i]);
}

// W[k][n] (KxN, fp32) -> Wt[n][k] (NxK, bf16) so WMMA B-fragments read contiguous K
__global__ void cvt_wt_kernel(const float* __restrict__ W,
                              uint16_t* __restrict__ Wt, int K, int N) {
  int i = blockIdx.x * blockDim.x + threadIdx.x;
  if (i >= K * N) return;
  int n = i % N, k = i / N;
  Wt[(size_t)n * K + k] = f2bf(W[i]);
}

// X: [B*T, NH*64] fp32 -> RoPE -> Y: [B, NH, T, 64] bf16
__global__ void rope_pack_kernel(const float* __restrict__ X,
                                 uint16_t* __restrict__ Y, int T, int NH) {
  int i = blockIdx.x * blockDim.x + threadIdx.x;
  int total = 2 * T * NH * 64;
  if (i >= total) return;
  int d   = i & 63;
  int r   = i >> 6;
  int hh  = r % NH;
  int row = r / NH;           // b*T + t
  int t = row % T, b = row / T;
  size_t base = (size_t)row * (NH * 64) + hh * 64;
  float x = X[base + d];
  float other = (d < 32) ? -X[base + d + 32] : X[base + d - 32];
  float ang = (float)t * __powf(10000.f, -(float)(d & 31) * (1.f / 32.f));
  float sn, cs;
  __sincosf(ang, &sn, &cs);
  Y[(((size_t)(b * NH + hh) * T) + t) * 64 + d] = f2bf(x * cs + other * sn);
}

// V: [B*T, 8*64] fp32 -> Y: [B, 8, 64, T] bf16 (transposed so P@V B-frags read contiguous K)
__global__ void v_pack_kernel(const float* __restrict__ V,
                              uint16_t* __restrict__ Y, int T) {
  int i = blockIdx.x * blockDim.x + threadIdx.x;
  int total = 2 * T * 8 * 64;
  if (i >= total) return;
  int d   = i & 63;
  int r   = i >> 6;
  int kv  = r % 8;
  int row = r / 8;
  int t = row % T, b = row / T;
  float v = V[(size_t)row * 512 + kv * 64 + d];
  Y[(((size_t)(b * 8 + kv) * 64) + d) * T + t] = f2bf(v);
}

// ---------------------------------------------------------------------------
// bf16 WMMA GEMM:  C[M,N] f32 = A[M,K]bf16 @ Bt[N,K]bf16^T + bias[N]
// Block tile 128x128, 8 waves, each wave 32x64 (2x4 WMMA 16x16 tiles), K-step 32.
// Double-buffered LDS, async global->LDS staging.
// ---------------------------------------------------------------------------
#define TILE_M 128
#define TILE_N 128
#define TILE_K 32

__global__ __launch_bounds__(256) void gemm_bf16_kernel(
    const uint16_t* __restrict__ A, const uint16_t* __restrict__ Bt,
    const float* __restrict__ bias, float* __restrict__ C,
    int M, int N, int K) {
  __shared__ uint16_t As[2][TILE_M][TILE_K];  // 2 x 8 KB
  __shared__ uint16_t Bs[2][TILE_N][TILE_K];  // 2 x 8 KB

  const int tid  = threadIdx.x;
  const int lane = tid & 31;
  const int wave = tid >> 5;
  const int wm   = wave & 3;   // 4 row-groups of 32
  const int wn   = wave >> 2;  // 2 col-groups of 64
  const int half = lane >> 4;  // 0/1
  const int ln   = lane & 15;

  const int mblk = blockIdx.y * TILE_M;
  const int nblk = blockIdx.x * TILE_N;

  v8f acc[2][4];
#pragma unroll
  for (int a = 0; a < 2; ++a)
#pragma unroll
    for (int b = 0; b < 4; ++b) acc[a][b] = (v8f)0.f;

  // stage one K-slab (A: 128x32, B: 128x32) = 2x512 16B chunks, 2+2 per thread
#define GEMM_STAGE(buf, k0)                                                     \
  do {                                                                          \
    _Pragma("unroll") for (int i_ = 0; i_ < 2; ++i_) {                          \
      int chunk_ = tid + i_ * 256;                                              \
      int row_ = chunk_ >> 2, cc_ = chunk_ & 3;                                 \
      async_copy16(A + (size_t)(mblk + row_) * K + (k0) + cc_ * 8,              \
                   &As[buf][row_][cc_ * 8]);                                    \
      async_copy16(Bt + (size_t)(nblk + row_) * K + (k0) + cc_ * 8,             \
                   &Bs[buf][row_][cc_ * 8]);                                    \
    }                                                                           \
  } while (0)

  GEMM_STAGE(0, 0);
  int cur = 0;

  for (int k0 = 0; k0 < K; k0 += TILE_K) {
    async_wait_all();
    __syncthreads();  // stage[cur] resident for all waves
    if (k0 + TILE_K < K) GEMM_STAGE(cur ^ 1, k0 + TILE_K);  // overlap next slab

    // A fragments (16x32, ISA 7.12.2 layout): lane = M, VGPR j -> K pair
    Frag16 af[2];
#pragma unroll
    for (int t = 0; t < 2; ++t) {
      const uint32_t* rp = (const uint32_t*)&As[cur][wm * 32 + t * 16 + ln][0];
#pragma unroll
      for (int j = 0; j < 8; ++j) {
        int kk = 2 * j + ((j >= 4) ? 8 : 0) + (half ? 8 : 0);
        af[t].u[j] = rp[kk >> 1];
      }
    }
    // B fragments (32x16): lane&15 = N, same K order as A fragments
    Frag16 bf[4];
#pragma unroll
    for (int t = 0; t < 4; ++t) {
      const uint32_t* rp = (const uint32_t*)&Bs[cur][wn * 64 + t * 16 + ln][0];
#pragma unroll
      for (int j = 0; j < 8; ++j) {
        int kk = 2 * j + ((j >= 4) ? 8 : 0) + (half ? 8 : 0);
        bf[t].u[j] = rp[kk >> 1];
      }
    }
#pragma unroll
    for (int a = 0; a < 2; ++a)
#pragma unroll
      for (int b = 0; b < 4; ++b)
        acc[a][b] = __builtin_amdgcn_wmma_f32_16x16x32_bf16(
            false, af[a].v, false, bf[b].v, (short)0, acc[a][b], false, false);
    __syncthreads();  // all reads of buf[cur] done before it is restaged
    cur ^= 1;
  }
#undef GEMM_STAGE

  // Epilogue: C layout -> M = j + 8*half, N = ln
#pragma unroll
  for (int a = 0; a < 2; ++a)
#pragma unroll
    for (int b = 0; b < 4; ++b) {
      int col = nblk + wn * 64 + b * 16 + ln;
      float bv = bias[col];
#pragma unroll
      for (int j = 0; j < 8; ++j) {
        int row = mblk + wm * 32 + a * 16 + 8 * half + j;
        C[(size_t)row * N + col] = acc[a][b][j] + bv;
      }
    }
}

// ---------------------------------------------------------------------------
// Flash attention (causal, GQA rep=2) with online softmax.
// Grid: (T/128, 16 heads, B). 8 waves, each owns 16 query rows, full head dim 64.
// K/V tiles staged into LDS once per block (double-buffered, async).
// qb:[B,16,T,64]  kb:[B,8,T,64]  vtb:[B,8,64,T]  -> ao:[B*T, 1024] bf16
// ---------------------------------------------------------------------------
#define FA_BR 128
#define FA_BC 32

__global__ __launch_bounds__(256) void flash_attn_kernel(
    const uint16_t* __restrict__ qb, const uint16_t* __restrict__ kb,
    const uint16_t* __restrict__ vtb, uint16_t* __restrict__ ao, int T) {
  __shared__ uint16_t Kt[2][FA_BC][64];   // 2 x 4 KB   [key][hd]
  __shared__ uint16_t Vt[2][64][FA_BC];   // 2 x 4 KB   [hd][key]
  __shared__ uint16_t Pls[8][16][FA_BC];  // per-wave P staging, 8 KB

  const int tid  = threadIdx.x;
  const int lane = tid & 31;
  const int wave = tid >> 5;
  const int half = lane >> 4;
  const int ln   = lane & 15;

  const int h    = blockIdx.y;
  const int b    = blockIdx.z;
  const int qblk = blockIdx.x * FA_BR;
  const int kv   = h >> 1;  // 16 Q heads -> 8 KV heads
  const int qrow0 = qblk + wave * 16;

  const uint16_t* qptr = qb  + (size_t)(b * 16 + h)  * T * 64;
  const uint16_t* kptr = kb  + (size_t)(b * 8  + kv) * T * 64;
  const uint16_t* vptr = vtb + (size_t)(b * 8  + kv) * 64 * T;

  // Q fragments, resident all kernel: two K-steps over head dim (0-31, 32-63)
  Frag16 qa[2];
#pragma unroll
  for (int t = 0; t < 2; ++t) {
    const uint32_t* rp = (const uint32_t*)(qptr + (size_t)(qrow0 + ln) * 64);
#pragma unroll
    for (int j = 0; j < 8; ++j) {
      int kk = t * 32 + 2 * j + ((j >= 4) ? 8 : 0) + (half ? 8 : 0);
      qa[t].u[j] = rp[kk >> 1];
    }
  }

  v8f o[4];
#pragma unroll
  for (int t = 0; t < 4; ++t) o[t] = (v8f)0.f;
  float mrow[8], lrow[8];
#pragma unroll
  for (int j = 0; j < 8; ++j) { mrow[j] = -1e30f; lrow[j] = 0.f; }

  const float scale = 0.125f;      // 1/sqrt(64)
  const int kend = qblk + FA_BR;   // block-uniform causal bound

  // stage 32-key K tile (32x64 = 256 chunks) + V tile (64x32 = 256 chunks)
#define FA_STAGE(buf, kb0)                                                      \
  do {                                                                          \
    int key_ = tid >> 3, ks_ = tid & 7;                                         \
    async_copy16(kptr + (size_t)((kb0) + key_) * 64 + ks_ * 8,                  \
                 &Kt[buf][key_][ks_ * 8]);                                      \
    int d_ = tid >> 2, vs_ = tid & 3;                                           \
    async_copy16(vptr + (size_t)d_ * T + (kb0) + vs_ * 8,                       \
                 &Vt[buf][d_][vs_ * 8]);                                        \
  } while (0)

  FA_STAGE(0, 0);
  int cur = 0;

  for (int kb0 = 0; kb0 < kend; kb0 += FA_BC) {
    async_wait_all();
    __syncthreads();  // K/V tile[cur] resident
    if (kb0 + FA_BC < kend) FA_STAGE(cur ^ 1, kb0 + FA_BC);  // overlap next tile

    // ---- S = q @ k^T (16 rows x 32 keys) : 2 tiles x 2 K-steps = 4 WMMAs ----
    v8f s[2];
#pragma unroll
    for (int t = 0; t < 2; ++t) {
      const uint32_t* rp = (const uint32_t*)&Kt[cur][t * 16 + ln][0];
      v8f c = (v8f)0.f;
#pragma unroll
      for (int ks = 0; ks < 2; ++ks) {
        Frag16 kf;
#pragma unroll
        for (int j = 0; j < 8; ++j) {
          int kk = ks * 32 + 2 * j + ((j >= 4) ? 8 : 0) + (half ? 8 : 0);
          kf.u[j] = rp[kk >> 1];
        }
        c = __builtin_amdgcn_wmma_f32_16x16x32_bf16(
            false, qa[ks].v, false, kf.v, (short)0, c, false, false);
      }
      s[t] = c;
    }

    // ---- scale, causal mask, online softmax (row = qrow0 + 8*half + j) ----
#pragma unroll
    for (int j = 0; j < 8; ++j) {
      int row = qrow0 + 8 * half + j;
      float v0 = s[0][j] * scale;
      float v1 = s[1][j] * scale;
      if (kb0 + ln > row)      v0 = -1e30f;
      if (kb0 + 16 + ln > row) v1 = -1e30f;
      float m = fmaxf(v0, v1);
#pragma unroll
      for (int off = 1; off < 16; off <<= 1)
        m = fmaxf(m, __shfl_xor(m, off, 32));
      float mnew  = fmaxf(mrow[j], m);
      float alpha = __expf(mrow[j] - mnew);
      float p0 = __expf(v0 - mnew);
      float p1 = __expf(v1 - mnew);
      float rs = p0 + p1;
#pragma unroll
      for (int off = 1; off < 16; off <<= 1)
        rs += __shfl_xor(rs, off, 32);
      lrow[j] = lrow[j] * alpha + rs;
      mrow[j] = mnew;
      o[0][j] *= alpha; o[1][j] *= alpha; o[2][j] *= alpha; o[3][j] *= alpha;
      Pls[wave][8 * half + j][ln]      = f2bf(p0);
      Pls[wave][8 * half + j][16 + ln] = f2bf(p1);
    }
    __syncthreads();  // uniform trip count across waves

    // ---- O += P @ V : re-read P in A layout, 4 WMMAs over head dim ----
    Frag16 pa;
    {
      const uint32_t* rp = (const uint32_t*)&Pls[wave][ln][0];
#pragma unroll
      for (int j = 0; j < 8; ++j) {
        int kk = 2 * j + ((j >= 4) ? 8 : 0) + (half ? 8 : 0);
        pa.u[j] = rp[kk >> 1];
      }
    }
#pragma unroll
    for (int t = 0; t < 4; ++t) {
      const uint32_t* rp = (const uint32_t*)&Vt[cur][t * 16 + ln][0];
      Frag16 vf;
#pragma unroll
      for (int j = 0; j < 8; ++j) {
        int kk = 2 * j + ((j >= 4) ? 8 : 0) + (half ? 8 : 0);
        vf.u[j] = rp[kk >> 1];
      }
      o[t] = __builtin_amdgcn_wmma_f32_16x16x32_bf16(
          false, pa.v, false, vf.v, (short)0, o[t], false, false);
    }
    __syncthreads();  // reads of tile[cur] done before restage
    cur ^= 1;
  }
#undef FA_STAGE

  // ---- normalize + pack to [B*T, H*64] bf16 ----
#pragma unroll
  for (int t = 0; t < 4; ++t) {
    int d = t * 16 + ln;
#pragma unroll
    for (int j = 0; j < 8; ++j) {
      int row = qrow0 + 8 * half + j;
      float val = o[t][j] / lrow[j];
      ao[(size_t)(b * T + row) * 1024 + h * 64 + d] = f2bf(val);
    }
  }
}

// ---------------------------------------------------------------------------
// Host-side orchestration
// ---------------------------------------------------------------------------
static inline uint8_t* carve(uint8_t*& p, size_t bytes) {
  uint8_t* r = p;
  p += (bytes + 255) & ~(size_t)255;
  return r;
}

extern "C" void kernel_launch(void* const* d_in, const int* in_sizes, int n_in,
                              void* d_out, int out_size, void* d_ws, size_t ws_size,
                              hipStream_t stream) {
  (void)in_sizes; (void)n_in; (void)out_size; (void)ws_size;
  const float* x  = (const float*)d_in[0];
  const float* Wq = (const float*)d_in[1];
  const float* bq = (const float*)d_in[2];
  const float* Wk = (const float*)d_in[3];
  const float* bk = (const float*)d_in[4];
  const float* Wv = (const float*)d_in[5];
  const float* bv = (const float*)d_in[6];
  const float* Wo = (const float*)d_in[7];
  const float* bo = (const float*)d_in[8];

  constexpr int B = 2, T = 2048, E = 1024, M = B * T;  // M = 4096

  uint8_t* p = (uint8_t*)d_ws;
  uint16_t* xb  = (uint16_t*)carve(p, (size_t)M * E * 2);        // x bf16
  uint16_t* wqt = (uint16_t*)carve(p, (size_t)E * E * 2);        // Wq^T bf16
  uint16_t* wkt = (uint16_t*)carve(p, (size_t)E * (E / 2) * 2);
  uint16_t* wvt = (uint16_t*)carve(p, (size_t)E * (E / 2) * 2);
  uint16_t* wot = (uint16_t*)carve(p, (size_t)E * E * 2);
  uint16_t* qbb = (uint16_t*)carve(p, (size_t)M * E * 2);        // [B,16,T,64]
  uint16_t* kbb = (uint16_t*)carve(p, (size_t)M * (E / 2) * 2);  // [B,8,T,64]
  uint16_t* vtb = (uint16_t*)carve(p, (size_t)M * (E / 2) * 2);  // [B,8,64,T]
  float*    qf  = (float*)carve(p, (size_t)M * E * 4);           // fp32 Q proj
  float*    kf  = (float*)carve(p, (size_t)M * (E / 2) * 4);
  float*    vf  = (float*)carve(p, (size_t)M * (E / 2) * 4);
  uint16_t* aob = (uint16_t*)qf;  // reuse qf space after RoPE consumes it

  const int TB = 256;
  // 1) conversions / weight transposes
  cvt_bf16_kernel<<<(M * E + TB - 1) / TB, TB, 0, stream>>>(x, xb, M * E);
  cvt_wt_kernel<<<(E * E + TB - 1) / TB, TB, 0, stream>>>(Wq, wqt, E, E);
  cvt_wt_kernel<<<(E * E / 2 + TB - 1) / TB, TB, 0, stream>>>(Wk, wkt, E, E / 2);
  cvt_wt_kernel<<<(E * E / 2 + TB - 1) / TB, TB, 0, stream>>>(Wv, wvt, E, E / 2);
  cvt_wt_kernel<<<(E * E + TB - 1) / TB, TB, 0, stream>>>(Wo, wot, E, E);

  // 2) QKV projections (WMMA GEMMs)
  gemm_bf16_kernel<<<dim3(E / TILE_N, M / TILE_M), TB, 0, stream>>>(
      xb, wqt, bq, qf, M, E, E);
  gemm_bf16_kernel<<<dim3(E / 2 / TILE_N, M / TILE_M), TB, 0, stream>>>(
      xb, wkt, bk, kf, M, E / 2, E);
  gemm_bf16_kernel<<<dim3(E / 2 / TILE_N, M / TILE_M), TB, 0, stream>>>(
      xb, wvt, bv, vf, M, E / 2, E);

  // 3) RoPE + layout packs (Q/K), V transpose pack
  rope_pack_kernel<<<(M * E + TB - 1) / TB, TB, 0, stream>>>(qf, qbb, T, 16);
  rope_pack_kernel<<<(M * E / 2 + TB - 1) / TB, TB, 0, stream>>>(kf, kbb, T, 8);
  v_pack_kernel<<<(M * E / 2 + TB - 1) / TB, TB, 0, stream>>>(vf, vtb, T);

  // 4) causal GQA flash attention (WMMA)
  flash_attn_kernel<<<dim3(T / FA_BR, 16, B), TB, 0, stream>>>(
      qbb, kbb, vtb, aob, T);

  // 5) output projection (WMMA GEMM) -> fp32 d_out
  gemm_bf16_kernel<<<dim3(E / TILE_N, M / TILE_M), TB, 0, stream>>>(
      aob, wot, bo, (float*)d_out, M, E, E);
}
